// KANBlock_49039936586417
// MI455X (gfx1250) — matrix operand order repaired and checked by reading, not compiled, not run on hardware
//
#include <hip/hip_runtime.h>
#include <hip/hip_bf16.h>

typedef __attribute__((ext_vector_type(16))) _Float16 v16h;
typedef __attribute__((ext_vector_type(8)))  _Float16 v8h;
typedef __attribute__((ext_vector_type(4)))  _Float16 v4h;
typedef __attribute__((ext_vector_type(8)))  float    v8f;

#define T_TOK   8192
#define D_IN    1024
#define H_HID   4096
#define LDA_PAD 40   // 32 halves + 8 pad (80B row stride, 16B aligned)

__device__ __forceinline__ float clipf(float v, float lo, float hi) {
    return fminf(fmaxf(v, lo), hi);
}

// ---------------------------------------------------------------------------
// Prep: clip coeff to [-1,1], convert to f16, transpose [O][k][Din] ->
// N-major B^T layout Bt[o][4*d + k] so GEMM B tiles are contiguous K rows.
// ---------------------------------------------------------------------------
__global__ void kan_prep_coeff(const float* __restrict__ c,
                               _Float16* __restrict__ bt,
                               int O, int Din) {
    size_t i = (size_t)blockIdx.x * blockDim.x + threadIdx.x;
    size_t total = (size_t)O * 4 * Din;
    if (i >= total) return;
    int din4 = 4 * Din;
    int o   = (int)(i / din4);
    int rem = (int)(i - (size_t)o * din4);
    int k   = rem / Din;
    int d   = rem - k * Din;
    float v = clipf(c[i], -1.0f, 1.0f);
    bt[(size_t)o * din4 + (size_t)d * 4 + k] = (_Float16)v;
}

// ---------------------------------------------------------------------------
// Tiled KAN GEMM: C[T, N] = PowA(src)[T, K] * Bt[N, K]^T   (K = 4 * KD)
//   Block tile 128(M) x 128(N), K-step 32. 8 waves: 4 in M x 2 in N,
//   each wave owns a 32x64 tile = 2x4 fragments of v_wmma_f32_16x16x32_f16.
// LAYER2=false: A from f32 x, epilogue = gelu(clip(0.1*acc)) -> f16 G
// LAYER2=true : A from f16 G, epilogue = clip(0.1*acc) + resid -> f32 Y
// ---------------------------------------------------------------------------
template <bool LAYER2>
__global__ __launch_bounds__(256)
void kan_gemm(const float* __restrict__ Xf,        // layer1 A source [T, KD]
              const _Float16* __restrict__ Gh,     // layer2 A source [T, KD]
              const _Float16* __restrict__ Bt,     // [N, K] f16, K-contiguous
              const float* __restrict__ resid,     // layer2 residual [T, N]
              _Float16* __restrict__ Gout,         // layer1 output [T, N] f16
              float* __restrict__ Yout,            // layer2 output [T, N] f32
              int KD, int K, int N) {
    __shared__ _Float16 As[128 * LDA_PAD];
    __shared__ _Float16 Bs[128 * LDA_PAD];

    const int tid   = threadIdx.x;
    const int lane  = tid & 31;
    const int wave  = tid >> 5;
    const int waveM = wave & 3;          // 4 waves in M
    const int waveN = wave >> 2;         // 2 waves in N
    const int hi    = lane >> 4;         // upper half-wave selector
    const int l16   = lane & 15;

    const int tBase = blockIdx.x * 128;  // M origin
    const int nBase = blockIdx.y * 128;  // N origin

    v8f acc[2][4];
#pragma unroll
    for (int im = 0; im < 2; ++im)
#pragma unroll
        for (int in = 0; in < 4; ++in)
            acc[im][in] = (v8f)(0.0f);

    union Frag { v16h v; v8h h[2]; };

    for (int ks = 0; ks < K; ks += 32) {
        __syncthreads();

        // ---- Build A tile in LDS: As[r][4*dg + k] = x[t][ks/4+dg]^k -------
#pragma unroll
        for (int it = 0; it < 4; ++it) {
            int idx = tid + it * 256;        // 0..1023
            int r   = idx >> 3;              // row in tile
            int dg  = idx & 7;               // d group (8 per K-step)
            int d   = (ks >> 2) + dg;
            float xv;
            if (LAYER2) xv = (float)Gh[(size_t)(tBase + r) * KD + d];
            else        xv = Xf[(size_t)(tBase + r) * KD + d];
            xv = clipf(xv, -5.0f, 5.0f);
            float x2 = xv * xv;
            float x3 = x2 * xv;
            v4h p = { (_Float16)1.0f, (_Float16)xv, (_Float16)x2, (_Float16)x3 };
            *(v4h*)(&As[r * LDA_PAD + dg * 4]) = p;
        }

        // ---- Stage B tile: Bs[n][0..31] = Bt[nBase+n][ks .. ks+32] --------
#pragma unroll
        for (int it = 0; it < 2; ++it) {
            int idx = tid + it * 256;        // 0..511
            int n   = idx >> 2;
            int seg = idx & 3;
            const _Float16* src = &Bt[(size_t)(nBase + n) * K + ks + seg * 8];
            *(v8h*)(&Bs[n * LDA_PAD + seg * 8]) = *(const v8h*)src;
            if (ks + 32 < K) __builtin_prefetch(src + 32, 0, 1);  // next K tile
        }

        __syncthreads();

        // ---- Load fragments (wave32 WMMA VGPR striping) -------------------
        // A frag lane L: row = L&15, K in {0..7,16..23}+8*(L>=16)
        Frag afr[2];
#pragma unroll
        for (int im = 0; im < 2; ++im) {
            int row = waveM * 32 + im * 16 + l16;
            afr[im].h[0] = *(const v8h*)(&As[row * LDA_PAD + 8 * hi]);
            afr[im].h[1] = *(const v8h*)(&As[row * LDA_PAD + 16 + 8 * hi]);
        }
        // B frag lane L: col = L&15, K in {0..7,16..23}+8*(L>=16)
        Frag bfr[4];
#pragma unroll
        for (int in = 0; in < 4; ++in) {
            int col = waveN * 64 + in * 16 + l16;
            bfr[in].h[0] = *(const v8h*)(&Bs[col * LDA_PAD + 8 * hi]);
            bfr[in].h[1] = *(const v8h*)(&Bs[col * LDA_PAD + 16 + 8 * hi]);
        }

#pragma unroll
        for (int im = 0; im < 2; ++im)
#pragma unroll
            for (int in = 0; in < 4; ++in)
                acc[im][in] = __builtin_amdgcn_wmma_f32_16x16x32_f16(
                    false, afr[im].v, false, bfr[in].v,
                    (short)0, acc[im][in], false, false);
    }

    // ---- Epilogue: C/D layout VGPR r, lane L -> M = r + 8*(L>=16), N = L&15
#pragma unroll
    for (int im = 0; im < 2; ++im) {
#pragma unroll
        for (int in = 0; in < 4; ++in) {
            int colBase = nBase + waveN * 64 + in * 16 + l16;
#pragma unroll
            for (int r = 0; r < 8; ++r) {
                int row = tBase + waveM * 32 + im * 16 + r + 8 * hi;
                float h = 0.1f * acc[im][in][r];
                h = clipf(h, -5.0f, 5.0f);
                if (LAYER2) {
                    float y = h + resid[(size_t)row * N + colBase];
                    Yout[(size_t)row * N + colBase] = y;
                } else {
                    // exact GELU: 0.5*h*(1+erf(h/sqrt(2)))
                    float g = 0.5f * h * (1.0f + erff(h * 0.70710678118654752f));
                    Gout[(size_t)row * N + colBase] = (_Float16)g;
                }
            }
        }
    }
}

// ---------------------------------------------------------------------------
// Row LayerNorm over D=1024, in place on y. One block (256 thr) per token.
// ---------------------------------------------------------------------------
__global__ __launch_bounds__(256)
void kan_layernorm(float* __restrict__ y,
                   const float* __restrict__ gamma,
                   const float* __restrict__ beta) {
    __shared__ float s1[256];
    __shared__ float s2[256];
    const int t   = blockIdx.x;
    const int tid = threadIdx.x;
    float v[4];
    float sum = 0.0f, sq = 0.0f;
#pragma unroll
    for (int i = 0; i < 4; ++i) {
        v[i] = y[(size_t)t * 1024 + tid + i * 256];
        sum += v[i];
        sq  += v[i] * v[i];
    }
    s1[tid] = sum;
    s2[tid] = sq;
    __syncthreads();
    for (int s = 128; s > 0; s >>= 1) {
        if (tid < s) { s1[tid] += s1[tid + s]; s2[tid] += s2[tid + s]; }
        __syncthreads();
    }
    const float mu  = s1[0] * (1.0f / 1024.0f);
    const float var = s2[0] * (1.0f / 1024.0f) - mu * mu;
    const float rs  = rsqrtf(var + 1e-5f);
#pragma unroll
    for (int i = 0; i < 4; ++i) {
        int n = tid + i * 256;
        y[(size_t)t * 1024 + n] = (v[i] - mu) * rs * gamma[n] + beta[n];
    }
}

// ---------------------------------------------------------------------------
extern "C" void kernel_launch(void* const* d_in, const int* in_sizes, int n_in,
                              void* d_out, int out_size, void* d_ws, size_t ws_size,
                              hipStream_t stream) {
    const float* x     = (const float*)d_in[0];   // [4,2048,1024]
    const float* c1    = (const float*)d_in[1];   // [4096,4,1024]
    const float* c2    = (const float*)d_in[2];   // [1024,4,4096]
    const float* gamma = (const float*)d_in[3];   // [1024]
    const float* beta  = (const float*)d_in[4];   // [1024]
    float* out = (float*)d_out;                   // [4,2048,1024]

    char* ws = (char*)d_ws;
    _Float16* B1t = (_Float16*)ws;                          // 32 MB: [4096][4096]
    _Float16* B2t = (_Float16*)(ws + ((size_t)32 << 20));   // 32 MB: [1024][16384]
    _Float16* G   = (_Float16*)(ws + ((size_t)64 << 20));   // 64 MB: [8192][4096]

    const int K1 = 4 * D_IN;    // 4096
    const int K2 = 4 * H_HID;   // 16384

    // coeff conversion: 16M elements each
    kan_prep_coeff<<<(H_HID * 4 * D_IN) / 256, 256, 0, stream>>>(c1, B1t, H_HID, D_IN);
    kan_prep_coeff<<<(D_IN * 4 * H_HID) / 256, 256, 0, stream>>>(c2, B2t, D_IN, H_HID);

    // layer 1: [8192 x 4096] = PowA(x)[8192 x 4096] * B1t^T, fused GELU
    kan_gemm<false><<<dim3(T_TOK / 128, H_HID / 128), 256, 0, stream>>>(
        x, nullptr, B1t, nullptr, G, nullptr, D_IN, K1, H_HID);

    // layer 2: [8192 x 1024] = PowA(G)[8192 x 16384] * B2t^T, fused residual
    kan_gemm<true><<<dim3(T_TOK / 128, D_IN / 128), 256, 0, stream>>>(
        nullptr, G, B2t, x, nullptr, out, H_HID, K2, D_IN);

    // in-place LayerNorm
    kan_layernorm<<<T_TOK, 256, 0, stream>>>(out, gamma, beta);
}
